// LSTM_19387482374377
// MI455X (gfx1250) — matrix operand order repaired and checked by reading, not compiled, not run on hardware
//
#include <hip/hip_runtime.h>
#include <hip/hip_bf16.h>

// ---------------------------------------------------------------------------
// LSTM on MI455X (gfx1250): persistent fused-GEMM recurrent kernel using
// v_wmma_f32_16x16x32_f16.  gates[t] = [x_t | h_{t-1}] @ [W_ih | W_hh]^T + b
//   M = 128 (batch), N = 4096 (4 gates x 1024), K = 1280 (256 + 1024)
// 64 WGs x 256 threads; WG w owns hidden columns 16w..16w+15 of every gate.
// Weight strip (~160 KB) staged once in LDS (CDNA5: 320 KB/WGP); inner loop
// reads B-tiles via ds_load_b128 with pure-immediate offsets (rows are
// interleaved col*4+gate so one base VGPR serves all 4 gates).
// One grid barrier per timestep; h ping-pong buffered.
// ---------------------------------------------------------------------------

typedef __attribute__((ext_vector_type(16))) _Float16 v16h;
typedef __attribute__((ext_vector_type(8)))  _Float16 v8h;
typedef __attribute__((ext_vector_type(8)))  float    v8f;

#define BATCH   128
#define SEQ     512
#define ISZ     256
#define HID     1024
#define KTOT    1280          // ISZ + HID
#define NWG     64
#define NTHR    256
// LDS row stride in halves: 1282 -> 641 dwords/row; lane stride = 4*RS =
// 2564 dwords === 4 banks (mod 64)  =>  16 column-lanes of a ds_load_b128
// land on disjoint 4-bank groups (conflict-free).
#define RS      1282

__device__ __forceinline__ float sigmf(float x) {
    return 1.0f / (1.0f + __expf(-x));
}

// --------------------------- prep kernels ----------------------------------

// xh[s][b][i] = (f16) x[b][s][i]
__global__ void k_conv_x(const float* __restrict__ x, _Float16* __restrict__ xh) {
    int idx = blockIdx.x * blockDim.x + threadIdx.x;   // SEQ*BATCH*ISZ = 16.7M
    if (idx >= SEQ * BATCH * ISZ) return;
    int i = idx % ISZ;
    int r = idx / ISZ;
    int b = r % BATCH;
    int s = r / BATCH;
    xh[((size_t)s * BATCH + b) * ISZ + i] = (_Float16)x[((size_t)b * SEQ + s) * ISZ + i];
}

// Wcat[n][k] fp16: k<256 -> W_ih[n][k], else W_hh[n][k-256]   (n = 0..4095)
__global__ void k_pack_w(const float* __restrict__ W_ih, const float* __restrict__ W_hh,
                         _Float16* __restrict__ Wcat) {
    int idx = blockIdx.x * blockDim.x + threadIdx.x;   // 4096*1280
    if (idx >= 4 * HID * KTOT) return;
    int k = idx % KTOT;
    int n = idx / KTOT;
    float v = (k < ISZ) ? W_ih[(size_t)n * ISZ + k] : W_hh[(size_t)n * HID + (k - ISZ)];
    Wcat[idx] = (_Float16)v;
}

// biasc[n] = b_ih[n]+b_hh[n]; zero hbuf (2 buffers), cbuf, barrier ints
__global__ void k_init(const float* __restrict__ b_ih, const float* __restrict__ b_hh,
                       float* __restrict__ biasc, _Float16* __restrict__ hbuf,
                       float* __restrict__ cbuf, int* __restrict__ bar) {
    int idx = blockIdx.x * blockDim.x + threadIdx.x;   // 262144 threads
    if (idx < 2 * BATCH * HID) hbuf[idx] = (_Float16)0.0f;
    if (idx < BATCH * HID)     cbuf[idx] = 0.0f;
    if (idx < 4 * HID)         biasc[idx] = b_ih[idx] + b_hh[idx];
    if (idx < 2)               bar[idx] = 0;
}

// ------------------------ persistent recurrent kernel ----------------------

__device__ __forceinline__ void grid_barrier(int* cnt, int* gen) {
    __syncthreads();
    if (threadIdx.x == 0) {
        int g = __hip_atomic_load(gen, __ATOMIC_ACQUIRE, __HIP_MEMORY_SCOPE_AGENT);
        int a = __hip_atomic_fetch_add(cnt, 1, __ATOMIC_ACQ_REL, __HIP_MEMORY_SCOPE_AGENT);
        if (a == NWG - 1) {
            __hip_atomic_store(cnt, 0, __ATOMIC_RELAXED, __HIP_MEMORY_SCOPE_AGENT);
            __hip_atomic_fetch_add(gen, 1, __ATOMIC_RELEASE, __HIP_MEMORY_SCOPE_AGENT);
        } else {
            while (__hip_atomic_load(gen, __ATOMIC_ACQUIRE, __HIP_MEMORY_SCOPE_AGENT) == g) {
                __builtin_amdgcn_s_sleep(2);
            }
        }
    }
    __syncthreads();
}

// Load one 16x16x32 B-tile (two 16-B LDS reads, immediate offsets only)
#define LDB(g, kk)                                                             \
    __builtin_shufflevector(*(const v8h*)&w_lds[wb + (g) * RS + (kk)],         \
                            *(const v8h*)&w_lds[wb + (g) * RS + (kk) + 8],     \
                            0,1,2,3,4,5,6,7,8,9,10,11,12,13,14,15)

__global__ __launch_bounds__(NTHR, 1)
void k_lstm_persist(const _Float16* __restrict__ xh,     // [SEQ][BATCH][ISZ] f16
                    const _Float16* __restrict__ Wcat,   // [4096][1280] f16
                    const float*    __restrict__ biasc,  // [4096]
                    _Float16* hbuf,                      // 2 x [BATCH][HID] f16
                    float*    cbuf,                      // [BATCH][HID] f32
                    _Float16* hs,                        // [SEQ][BATCH][HID] f16
                    int* bar)
{
    // 64 rows x RS halves; row r = col*4 + gate  (col 0..15, gate 0..3)
    // 64 * 1282 * 2 = 164,096 B of the 320 KB WGP LDS
    __shared__ _Float16 w_lds[64 * RS];

    const int lane  = threadIdx.x & 31;
    const int wave  = threadIdx.x >> 5;        // 0..7  -> M-tile (batch rows)
    const int jb    = blockIdx.x * 16;         // hidden-column strip of this WG
    const int l15   = lane & 15;
    const bool lo16 = (lane < 16);
    const int jcol  = jb + l15;                // this lane's hidden column
    const int rb    = wave * 16;               // batch-row base of this wave
    // A-tile addressing (16-bit A 16x32, same k order as B): row = lane&15;
    // lanes 0-15 take K+0..15, lanes 16-31 take K+16..31 (contiguous)
    const int arow  = rb + l15;
    const int akA   = lo16 ? 0 : 16;
    const int akB   = lo16 ? 8 : 24;
    // B-tile addressing: column = lane&15, K-half by lane group (contig 16 halves)
    const int bk    = lo16 ? 0 : 16;

    // ---- stage this WG's weight strip into LDS (once per launch) ----------
    // LDS row r (0..63): col = r>>2, gate = r&3   -> n = (r&3)*HID + jb + (r>>2)
    for (int t = threadIdx.x; t < 64 * (KTOT / 8); t += NTHR) {
        int r  = t / (KTOT / 8);
        int kc = (t % (KTOT / 8)) * 8;
        int n  = (r & 3) * HID + jb + (r >> 2);
        v8h v = *(const v8h*)(Wcat + (size_t)n * KTOT + kc);
        *(v8h*)&w_lds[r * RS + kc] = v;
    }
    __syncthreads();

    const float bi = biasc[          jcol];
    const float bf = biasc[    HID + jcol];
    const float bg = biasc[2 * HID + jcol];
    const float bo = biasc[3 * HID + jcol];

    int* cnt = bar;
    int* gen = bar + 1;

    for (int s = 0; s < SEQ; ++s) {
        // Opaque zero: makes LDS B-tile addresses formally loop-variant so
        // LICM cannot hoist 160 KB of weights into VGPRs (round-1 spill bug).
        int wbase = 0;
        asm volatile("" : "+v"(wbase));
        // single per-lane LDS base; all 4 gates reachable via imm offsets
        const int wb = (l15 * 4) * RS + bk + wbase;

        const _Float16* hin  = hbuf + (size_t)(s & 1)       * (BATCH * HID);
        _Float16*       hout = hbuf + (size_t)((s + 1) & 1) * (BATCH * HID);
        const _Float16* xrow = xh + ((size_t)s * BATCH + arow) * ISZ;
        const _Float16* hrow = hin + (size_t)arow * HID;

        v8f a0 = {}, a1 = {}, a2 = {}, a3 = {};

        // ---- K over x part (k = 0..255, 8 wmma k-steps) -------------------
        #pragma unroll 4
        for (int kk = 0; kk < ISZ; kk += 32) {
            v8h pA = *(const v8h*)(xrow + kk + akA);
            v8h pB = *(const v8h*)(xrow + kk + akB);
            v16h a = __builtin_shufflevector(pA, pB, 0,1,2,3,4,5,6,7,8,9,10,11,12,13,14,15);
            v16h b0 = LDB(0, kk);
            v16h b1 = LDB(1, kk);
            v16h b2 = LDB(2, kk);
            v16h b3 = LDB(3, kk);
            a0 = __builtin_amdgcn_wmma_f32_16x16x32_f16(false, a, false, b0, (short)0, a0, false, false);
            a1 = __builtin_amdgcn_wmma_f32_16x16x32_f16(false, a, false, b1, (short)0, a1, false, false);
            a2 = __builtin_amdgcn_wmma_f32_16x16x32_f16(false, a, false, b2, (short)0, a2, false, false);
            a3 = __builtin_amdgcn_wmma_f32_16x16x32_f16(false, a, false, b3, (short)0, a3, false, false);
        }
        // ---- K over h part (k = 256..1279, 32 wmma k-steps) ---------------
        #pragma unroll 4
        for (int kk = 0; kk < HID; kk += 32) {
            v8h pA = *(const v8h*)(hrow + kk + akA);
            v8h pB = *(const v8h*)(hrow + kk + akB);
            v16h a = __builtin_shufflevector(pA, pB, 0,1,2,3,4,5,6,7,8,9,10,11,12,13,14,15);
            v16h b0 = LDB(0, ISZ + kk);
            v16h b1 = LDB(1, ISZ + kk);
            v16h b2 = LDB(2, ISZ + kk);
            v16h b3 = LDB(3, ISZ + kk);
            a0 = __builtin_amdgcn_wmma_f32_16x16x32_f16(false, a, false, b0, (short)0, a0, false, false);
            a1 = __builtin_amdgcn_wmma_f32_16x16x32_f16(false, a, false, b1, (short)0, a1, false, false);
            a2 = __builtin_amdgcn_wmma_f32_16x16x32_f16(false, a, false, b2, (short)0, a2, false, false);
            a3 = __builtin_amdgcn_wmma_f32_16x16x32_f16(false, a, false, b3, (short)0, a3, false, false);
        }

        // ---- elementwise LSTM cell update for this wave's 16x16 h tile ----
        // C/D layout: lanes 0-15: (M=v, N=lane); lanes 16-31: (M=v+8, N=lane-16)
        #pragma unroll
        for (int v = 0; v < 8; ++v) {
            int b = rb + v + (lo16 ? 0 : 8);
            size_t cidx = (size_t)b * HID + jcol;
            float iv = sigmf(a0[v] + bi);
            float fv = sigmf(a1[v] + bf);
            float gv = tanhf(a2[v] + bg);
            float ov = sigmf(a3[v] + bo);
            float cc = fv * cbuf[cidx] + iv * gv;
            cbuf[cidx] = cc;
            _Float16 hh = (_Float16)(ov * tanhf(cc));
            hout[cidx] = hh;
            hs[((size_t)s * BATCH + b) * HID + jcol] = hh;
        }

        __threadfence();            // make hout visible device-wide
        grid_barrier(cnt, gen);     // all WGs advance to step s+1 together
    }
}

// ------------------------- output projection -------------------------------
// out[b][s][o] = b_out[o] + sum_j hs[s][b][j] * W_out[o][j]   (OUT = 3)
// one wave per (b,s) position, shuffle reduction.
__global__ void k_out_proj(const _Float16* __restrict__ hs,
                           const float* __restrict__ W_out,
                           const float* __restrict__ b_out,
                           float* __restrict__ out) {
    int gw   = (blockIdx.x * blockDim.x + threadIdx.x) >> 5;
    int lane = threadIdx.x & 31;
    if (gw >= BATCH * SEQ) return;
    int s = gw % SEQ;
    int b = gw / SEQ;
    const _Float16* hp = hs + ((size_t)s * BATCH + b) * HID;
    float a0 = 0.f, a1 = 0.f, a2 = 0.f;
    for (int j = lane; j < HID; j += 32) {
        float hv = (float)hp[j];
        a0 += hv * W_out[j];
        a1 += hv * W_out[HID + j];
        a2 += hv * W_out[2 * HID + j];
    }
    for (int off = 16; off > 0; off >>= 1) {
        a0 += __shfl_xor(a0, off, 32);
        a1 += __shfl_xor(a1, off, 32);
        a2 += __shfl_xor(a2, off, 32);
    }
    if (lane == 0) {
        float* op = out + ((size_t)b * SEQ + s) * 3;
        op[0] = a0 + b_out[0];
        op[1] = a1 + b_out[1];
        op[2] = a2 + b_out[2];
    }
}

// ------------------------------ launch -------------------------------------

extern "C" void kernel_launch(void* const* d_in, const int* in_sizes, int n_in,
                              void* d_out, int out_size, void* d_ws, size_t ws_size,
                              hipStream_t stream) {
    const float* x     = (const float*)d_in[0];
    const float* W_ih  = (const float*)d_in[1];
    const float* W_hh  = (const float*)d_in[2];
    const float* b_ih  = (const float*)d_in[3];
    const float* b_hh  = (const float*)d_in[4];
    const float* W_out = (const float*)d_in[5];
    const float* b_out = (const float*)d_in[6];
    float* out = (float*)d_out;

    // workspace carve-up (256B aligned chunks)
    char* ws = (char*)d_ws;
    size_t off = 0;
    _Float16* xh    = (_Float16*)(ws + off); off += (size_t)SEQ * BATCH * ISZ * 2;      // 32 MB
    _Float16* Wcat  = (_Float16*)(ws + off); off += (size_t)4 * HID * KTOT * 2;         // 10 MB
    float*    biasc = (float*)   (ws + off); off += (size_t)4 * HID * 4;                // 16 KB
    _Float16* hbuf  = (_Float16*)(ws + off); off += (size_t)2 * BATCH * HID * 2;        // 512 KB
    float*    cbuf  = (float*)   (ws + off); off += (size_t)BATCH * HID * 4;            // 512 KB
    int*      bar   = (int*)     (ws + off); off += 256;
    _Float16* hs    = (_Float16*)(ws + off); off += (size_t)SEQ * BATCH * HID * 2;      // 128 MB

    // prep
    k_conv_x<<<(SEQ * BATCH * ISZ + 255) / 256, 256, 0, stream>>>(x, xh);
    k_pack_w<<<(4 * HID * KTOT + 255) / 256, 256, 0, stream>>>(W_ih, W_hh, Wcat);
    k_init<<<(2 * BATCH * HID + 255) / 256, 256, 0, stream>>>(b_ih, b_hh, biasc, hbuf, cbuf, bar);

    // recurrent persistent kernel: 64 co-resident WGs, 512 steps
    k_lstm_persist<<<NWG, NTHR, 0, stream>>>(xh, Wcat, biasc, hbuf, cbuf, hs, bar);

    // output projection: one wave per (b,s)
    k_out_proj<<<(BATCH * SEQ * 32 + 255) / 256, 256, 0, stream>>>(hs, W_out, b_out, out);
}
